// SGATEmbedding_67680094650926
// MI455X (gfx1250) — matrix operand order repaired and checked by reading, not compiled, not run on hardware
//
#include <hip/hip_runtime.h>

// ---- problem constants (from reference setup_inputs) ----
#define B_   8
#define T_   12
#define N_   10000
#define F_   16
#define E_   80000
#define D_   192            // F*T = in/out feature dim
#define V_   (B_*N_)        // 80000 batched nodes
#define EB_  (B_*E_)        // 640000 batched edges
#define ALPHA_ 0.2f

typedef __attribute__((ext_vector_type(16))) __bf16        v16bf;
typedef __attribute__((ext_vector_type(8)))  float         v8f;
typedef __attribute__((ext_vector_type(4)))  unsigned int  u32x4;

union BfTile { u32x4 u[2]; v16bf v; };

__device__ __forceinline__ unsigned short f2bf(float f) {
  unsigned int u = __float_as_uint(f);
  u += 0x7FFFu + ((u >> 16) & 1u);          // round-to-nearest-even
  return (unsigned short)(u >> 16);
}

// monotonic-encoding float atomic max (init must be -inf)
__device__ __forceinline__ void atomicMaxF(float* addr, float val) {
  if (val >= 0.f) atomicMax((int*)addr, __float_as_int(val));
  else            atomicMin((unsigned int*)addr, __float_as_uint(val));
}

// ---- x [B,T,N,F] f32  ->  h_bf16 [V, D] with h[b*N+n, f*T+t] = x[b,t,n,f] ----
__global__ void k_prep_x(const float* __restrict__ x, unsigned short* __restrict__ hb) {
  int id = blockIdx.x * blockDim.x + threadIdx.x;
  if (id >= B_*T_*N_*F_) return;
  int f  = id & (F_ - 1);
  int n  = (id >> 4) % N_;
  int bt = id / (F_ * N_);
  int t  = bt % T_;
  int b  = bt / T_;
  hb[(size_t)(b*N_ + n)*D_ + f*T_ + t] = f2bf(x[id]);
}

// ---- W[k][n] f32 -> Wt_bf16[n][k] (transposed, so B-fragment K is contiguous) ----
__global__ void k_prep_w(const float* __restrict__ W0, const float* __restrict__ W1,
                         unsigned short* __restrict__ Wt0, unsigned short* __restrict__ Wt1) {
  int id = blockIdx.x * blockDim.x + threadIdx.x;
  if (id >= 2*D_*D_) return;
  int m = id / (D_*D_);
  int r = id - m*D_*D_;
  int k = r / D_, n = r % D_;
  const float* W = m ? W1 : W0;
  unsigned short* Wt = m ? Wt1 : Wt0;
  Wt[n*D_ + k] = f2bf(W[k*D_ + n]);
}

__global__ void k_init(float* __restrict__ rst, float* __restrict__ emax,
                       float* __restrict__ denom) {
  int id = blockIdx.x * blockDim.x + threadIdx.x;
  if (id >= V_*D_) return;
  rst[id] = 0.f;
  if (id < V_) { emax[id] = -__builtin_inff(); denom[id] = 0.f; }
}

// ---- ft = h @ W via v_wmma_f32_16x16x32_bf16; fused el/er epilogue ----
// wave = one 16-row M-tile x full 192 cols (12 accumulators); block = 8 waves = 128 rows
__global__ void __launch_bounds__(256)
k_gemm_wmma(const unsigned short* __restrict__ A,   // [V,192] bf16 row-major
            const unsigned short* __restrict__ Bt,  // [192,192] bf16, Bt[n][k]
            const float* __restrict__ attn_l, const float* __restrict__ attn_r,
            float* __restrict__ C, float* __restrict__ el, float* __restrict__ er,
            int V) {
  const int lane = threadIdx.x & 31;
  const int wave = threadIdx.x >> 5;
  const int g    = lane >> 4;      // half-wave group
  const int lr   = lane & 15;
  const int rowBase = (blockIdx.x * 8 + wave) * 16;
  if (rowBase >= V) return;
  const int row = rowBase + lr;

  v8f acc[12];
  const v8f zero = {0.f,0.f,0.f,0.f,0.f,0.f,0.f,0.f};
  #pragma unroll
  for (int j = 0; j < 12; ++j) acc[j] = zero;

  const u32x4* Arow = (const u32x4*)(A + (size_t)row * D_);
  for (int kt = 0; kt < 6; ++kt) {
    BfTile a;                                  // A 16x32 frag: halves K = 16g..16g+15
    a.u[0] = Arow[kt*4 + 2*g];
    a.u[1] = Arow[kt*4 + 2*g + 1];
    #pragma unroll
    for (int j = 0; j < 12; ++j) {
      const u32x4* Bcol = (const u32x4*)(Bt + (size_t)(j*16 + lr)*D_ + kt*32 + 16*g);
      BfTile b;                                // B 32x16 frag: halves K = 16g..16g+15
      b.u[0] = Bcol[0];
      b.u[1] = Bcol[1];
      acc[j] = __builtin_amdgcn_wmma_f32_16x16x32_bf16(
          false, a.v, false, b.v, (short)0, acc[j], false, false);
    }
  }

  // epilogue: store ft, fuse el = ft . attn_l, er = ft . attn_r
  float pl[8], pr[8];
  #pragma unroll
  for (int r = 0; r < 8; ++r) { pl[r] = 0.f; pr[r] = 0.f; }
  #pragma unroll
  for (int j = 0; j < 12; ++j) {
    int col = j*16 + lr;
    float alv = attn_l[col], arv = attn_r[col];
    #pragma unroll
    for (int r = 0; r < 8; ++r) {
      float v = acc[j][r];                       // element (M=r+8g, N=col)
      C[(size_t)(rowBase + r + 8*g)*D_ + col] = v;
      pl[r] += v * alv;  pr[r] += v * arv;
    }
  }
  #pragma unroll
  for (int r = 0; r < 8; ++r) {
    #pragma unroll
    for (int m = 8; m >= 1; m >>= 1) {           // reduce over N within 16-lane group
      pl[r] += __shfl_xor(pl[r], m, 16);
      pr[r] += __shfl_xor(pr[r], m, 16);
    }
    if (lr == 0) {
      el[rowBase + r + 8*g] = pl[r];
      er[rowBase + r + 8*g] = pr[r];
    }
  }
}

// ---- edge logits + segment max over dst ----
__global__ void k_edge_logits(const float* __restrict__ el, const float* __restrict__ er,
                              const int* __restrict__ src, const int* __restrict__ dst,
                              float* __restrict__ ebuf, float* __restrict__ emax) {
  int id = blockIdx.x * blockDim.x + threadIdx.x;
  if (id >= EB_) return;
  int b = id / E_, e = id - b*E_;
  int s = src[e] + b*N_, d = dst[e] + b*N_;
  float v = el[s] + er[d];
  v = v > 0.f ? v : ALPHA_ * v;                  // leaky_relu
  ebuf[id] = v;
  atomicMaxF(&emax[d], v);
}

// ---- exp(e - max) + segment sum ----
__global__ void k_edge_exp(const int* __restrict__ dst, float* __restrict__ ebuf,
                           const float* __restrict__ emax, float* __restrict__ denom) {
  int id = blockIdx.x * blockDim.x + threadIdx.x;
  if (id >= EB_) return;
  int b = id / E_, e = id - b*E_;
  int d = dst[e] + b*N_;
  float ex = __expf(ebuf[id] - emax[d]);
  ebuf[id] = ex;
  atomicAdd(&denom[d], ex);
}

// ---- rst[dst] += (ex/denom[dst]) * ft[src]; one wave per edge, 6 f32 atomics/lane ----
__global__ void k_aggregate(const float* __restrict__ ebuf, const float* __restrict__ denom,
                            const int* __restrict__ src, const int* __restrict__ dst,
                            const float* __restrict__ ft, float* __restrict__ rst) {
  int eb   = blockIdx.x * 8 + (threadIdx.x >> 5);
  int lane = threadIdx.x & 31;
  if (eb >= EB_) return;
  int b = eb / E_, e = eb - b*E_;
  int s = src[e] + b*N_, d = dst[e] + b*N_;
  float a = ebuf[eb] / denom[d];
  const float* fs = ft + (size_t)s * D_;
  float* rd = rst + (size_t)d * D_;
  #pragma unroll
  for (int i = 0; i < 6; ++i) {
    int c = lane + i*32;
    atomicAdd(&rd[c], a * fs[c]);
  }
}

// ---- layer-1 epilogue: elu(rst + bias) -> bf16 input for layer 2 ----
__global__ void k_node_epilogue(const float* __restrict__ rst, const float* __restrict__ bias,
                                unsigned short* __restrict__ hb) {
  int id = blockIdx.x * blockDim.x + threadIdx.x;
  if (id >= V_*D_) return;
  int c = id % D_;
  float v = rst[id] + bias[c];
  v = v > 0.f ? v : (__expf(v) - 1.f);           // elu
  hb[id] = f2bf(v);
}

// ---- final: out[b,t,n,f] = x + elu(rst2 + bias1) at (b*N+n, f*T+t) ----
__global__ void k_final(const float* __restrict__ x, const float* __restrict__ rst,
                        const float* __restrict__ bias, float* __restrict__ out) {
  int id = blockIdx.x * blockDim.x + threadIdx.x;
  if (id >= B_*T_*N_*F_) return;
  int f  = id & (F_ - 1);
  int n  = (id >> 4) % N_;
  int bt = id / (F_ * N_);
  int t  = bt % T_;
  int b  = bt / T_;
  int c  = f*T_ + t;
  float v = rst[(size_t)(b*N_ + n)*D_ + c] + bias[c];
  v = v > 0.f ? v : (__expf(v) - 1.f);
  out[id] = x[id] + v;
}

extern "C" void kernel_launch(void* const* d_in, const int* in_sizes, int n_in,
                              void* d_out, int out_size, void* d_ws, size_t ws_size,
                              hipStream_t stream) {
  (void)in_sizes; (void)n_in; (void)out_size; (void)ws_size;
  const float* x     = (const float*)d_in[0];
  const int*   src   = (const int*)  d_in[1];
  const int*   dst   = (const int*)  d_in[2];
  const float* W0    = (const float*)d_in[3];
  const float* al0   = (const float*)d_in[4];
  const float* ar0   = (const float*)d_in[5];
  const float* bias0 = (const float*)d_in[6];
  const float* W1    = (const float*)d_in[7];
  const float* al1   = (const float*)d_in[8];
  const float* ar1   = (const float*)d_in[9];
  const float* bias1 = (const float*)d_in[10];
  float* out = (float*)d_out;

  // workspace carve-out (~158 MB total; everything L2-resident on a 192MB L2)
  char* w = (char*)d_ws;
  size_t off = 0;
  auto alloc = [&](size_t bytes) -> void* {
    void* p = w + off; off += (bytes + 255) & ~size_t(255); return p;
  };
  unsigned short* hb  = (unsigned short*)alloc((size_t)V_ * D_ * 2);
  unsigned short* Wt0 = (unsigned short*)alloc((size_t)D_ * D_ * 2);
  unsigned short* Wt1 = (unsigned short*)alloc((size_t)D_ * D_ * 2);
  float* ft    = (float*)alloc((size_t)V_ * D_ * 4);
  float* rst   = (float*)alloc((size_t)V_ * D_ * 4);
  float* el    = (float*)alloc((size_t)V_ * 4);
  float* er    = (float*)alloc((size_t)V_ * 4);
  float* emax  = (float*)alloc((size_t)V_ * 4);
  float* denom = (float*)alloc((size_t)V_ * 4);
  float* ebuf  = (float*)alloc((size_t)EB_ * 4);

  const int TB = 256;
  const int nx = B_*T_*N_*F_;
  k_prep_x<<<(nx + TB - 1)/TB, TB, 0, stream>>>(x, hb);
  k_prep_w<<<(2*D_*D_ + TB - 1)/TB, TB, 0, stream>>>(W0, W1, Wt0, Wt1);

  for (int layer = 0; layer < 2; ++layer) {
    const unsigned short* Wt = layer ? Wt1 : Wt0;
    const float* al = layer ? al1 : al0;
    const float* ar = layer ? ar1 : ar0;
    k_init<<<(V_*D_ + TB - 1)/TB, TB, 0, stream>>>(rst, emax, denom);
    k_gemm_wmma<<<(V_ + 127)/128, TB, 0, stream>>>(hb, Wt, al, ar, ft, el, er, V_);
    k_edge_logits<<<(EB_ + TB - 1)/TB, TB, 0, stream>>>(el, er, src, dst, ebuf, emax);
    k_edge_exp<<<(EB_ + TB - 1)/TB, TB, 0, stream>>>(dst, ebuf, emax, denom);
    k_aggregate<<<(EB_ + 7)/8, TB, 0, stream>>>(ebuf, denom, src, dst, ft, rst);
    if (layer == 0)
      k_node_epilogue<<<(V_*D_ + TB - 1)/TB, TB, 0, stream>>>(rst, bias0, hb);
    else
      k_final<<<(nx + TB - 1)/TB, TB, 0, stream>>>(x, rst, bias1, out);
  }
}
